// Net_GCN_48790828482988
// MI455X (gfx1250) — matrix operand order repaired and checked by reading, not compiled, not run on hardware
//
#include <hip/hip_runtime.h>

// ---------------------------------------------------------------------------
// Types for CDNA5 WMMA (wave32, 16x16x32 bf16 -> f32)
// ---------------------------------------------------------------------------
typedef __attribute__((ext_vector_type(16))) __bf16          v16bf;
typedef __attribute__((ext_vector_type(16))) unsigned short  v16u;
typedef __attribute__((ext_vector_type(8)))  unsigned short  u16x8;
typedef __attribute__((ext_vector_type(8)))  float           v8f;
typedef unsigned short u16;

// GCC-style v4i with explicit address spaces, matching the async-to-LDS
// builtin's parameter types (AS1 = global/"__device__", AS3 = LDS/"__shared__").
typedef int v4i_gcc __attribute__((__vector_size__(16)));
typedef __attribute__((address_space(1))) v4i_gcc gv4i;
typedef __attribute__((address_space(3))) v4i_gcc lv4i;

// f32 -> bf16 bits, round-to-nearest-even
__device__ __forceinline__ u16 f32_to_bf16_bits(float f) {
    unsigned int u = __float_as_uint(f);
    u += 0x7fffu + ((u >> 16) & 1u);
    return (u16)(u >> 16);
}

// ---------------------------------------------------------------------------
// Async 16B global -> LDS staging (gfx1250 GLOBAL_LOAD_ASYNC_TO_LDS_B128).
// Generic pointers to LDS carry the LDS byte offset in their low 32 bits
// (ISA 10.2 aperture rules), so inttoptr of the truncated address yields a
// valid AS3 pointer.
// ---------------------------------------------------------------------------
__device__ __forceinline__ void stage16_async(const u16* __restrict__ g, u16* l) {
#if __has_builtin(__builtin_amdgcn_global_load_async_to_lds_b128)
    __builtin_amdgcn_global_load_async_to_lds_b128(
        (gv4i*)(uintptr_t)g,
        (lv4i*)(unsigned int)(uintptr_t)l,
        0, 0);
#else
    unsigned int       loff = (unsigned int)(uintptr_t)l;
    unsigned long long ga   = (unsigned long long)(uintptr_t)g;
    asm volatile("global_load_async_to_lds_b128 %0, %1, off"
                 :: "v"(loff), "v"(ga) : "memory");
#endif
}

__device__ __forceinline__ void async_wait_all() {
#if __has_builtin(__builtin_amdgcn_s_wait_asynccnt)
    __builtin_amdgcn_s_wait_asynccnt(0);
#else
    asm volatile("s_wait_asynccnt 0" ::: "memory");
#endif
}

// ---------------------------------------------------------------------------
// deg[i] = 1.0 (self-loop weight)
// ---------------------------------------------------------------------------
__global__ void init_deg_kernel(float* __restrict__ deg, int n) {
    int i = blockIdx.x * blockDim.x + threadIdx.x;
    if (i < n) deg[i] = 1.0f;
}

__global__ void accum_deg_kernel(const int* __restrict__ dst,
                                 const float* __restrict__ ew,
                                 float* __restrict__ deg, int E) {
    int e = blockIdx.x * blockDim.x + threadIdx.x;
    if (e < E) atomicAdd(&deg[dst[e]], ew[e]);
}

__global__ void finalize_dinv_kernel(float* __restrict__ deg, int n) {
    int i = blockIdx.x * blockDim.x + threadIdx.x;
    if (i < n) {
        float d = deg[i];
        deg[i] = (d > 0.0f) ? rsqrtf(d) : 0.0f;
    }
}

// ---------------------------------------------------------------------------
// f32 -> bf16 (RNE), optional fused ReLU (for activations / A matrices)
// ---------------------------------------------------------------------------
__global__ void f32_to_bf16_kernel(const float* __restrict__ in,
                                   u16* __restrict__ out,
                                   size_t n, int relu) {
    size_t i = (size_t)blockIdx.x * blockDim.x + threadIdx.x;
    if (i >= n) return;
    float f = in[i];
    if (relu) f = f > 0.0f ? f : 0.0f;
    out[i] = f32_to_bf16_bits(f);
}

// ---------------------------------------------------------------------------
// Pack W[K,N] (f32 row-major) into WMMA B-fragment-major bf16:
//   Bp[(((tn*(K/32) + kb32)*32 + lane)*16 + i]
// where lane = half*16+lrow, fragment element i: (v = i>>1, p = i&1)
//   K-index = kb32*32 + half*16 + 2v + p,  N-index = tn*16 + lrow.
// Each lane's 16-element fragment is 32 contiguous bytes -> 2x b128 loads.
// ---------------------------------------------------------------------------
__global__ void pack_B_kernel(const float* __restrict__ W,
                              u16* __restrict__ Bp, int N, int K) {
    const int nkb = K >> 5;
    size_t gid = (size_t)blockIdx.x * blockDim.x + threadIdx.x;
    if (gid >= (size_t)K * N) return;
    const int i    = (int)(gid & 15);
    const int l    = (int)((gid >> 4) & 31);
    const int t    = (int)(gid >> 9);          // tn*nkb + kb32
    const int kb32 = t % nkb;
    const int tn   = t / nkb;
    const int half = l >> 4, lrow = l & 15;
    const int kk   = kb32 * 32 + half * 16 + 2 * (i >> 1) + (i & 1);
    const int nn   = tn * 16 + lrow;
    Bp[gid] = f32_to_bf16_bits(W[(size_t)kk * N + nn]);
}

// ---------------------------------------------------------------------------
// C[M,N] = A[M,K] @ B[K,N]; A bf16 row-major, Bp packed fragment-major bf16,
// N = NT*16 (compile time). Whole packed B staged to LDS via async-to-LDS;
// one wave computes a full 16xN output strip (NT accumulators), so each A
// fragment is loaded exactly once. K and N-tiles fully unrolled:
// NT*K/32 straight-line v_wmma_f32_16x16x32_bf16 per wave.
// Requires M%16==0 (true: M=100000).
// ---------------------------------------------------------------------------
template <int K, int NT>
__global__ __launch_bounds__(256)
void gemm_bf16_wmma_kernel(const u16* __restrict__ A,
                           const u16* __restrict__ Bp,
                           float* __restrict__ C,
                           int M) {
    constexpr int NKB = K / 32;
    constexpr int N   = NT * 16;
    __shared__ u16 ldsB[N * K];                // 32 KB (NT=8) / 16 KB (NT=4)
    const int tid = threadIdx.x;

    // Cooperative async stage of packed B into LDS (16 B per op).
    constexpr int nchunks = (N * K) / 8;       // # of 8-u16 chunks
    for (int c = tid; c < nchunks; c += 256)
        stage16_async(Bp + (size_t)c * 8, ldsB + (size_t)c * 8);
    async_wait_all();
    __syncthreads();

    const int mtiles = M >> 4;
    const int wrow   = blockIdx.x * 8 + (tid >> 5);  // 16-row strip per wave
    if (wrow >= mtiles) return;                // wave-uniform, after barrier
    const int lane = tid & 31;
    const int half = lane >> 4;
    const int lrow = lane & 15;

    v8f acc[NT];
    #pragma unroll
    for (int nt = 0; nt < NT; ++nt) acc[nt] = (v8f){};

    const u16* __restrict__ arow  = A + (size_t)(wrow * 16 + lrow) * K + half * 16;
    const u16* __restrict__ bbase = ldsB + (size_t)lane * 16;

    #pragma unroll
    for (int kb32 = 0; kb32 < NKB; ++kb32) {
        const int kb = kb32 * 32;
        // A 16x32 bf16: lane(M=lrow) holds K = kb+half*16+j (elems 0..15):
        // two contiguous 16B loads.
        u16x8 alo = *(const u16x8*)(arow + kb);
        u16x8 ahi = *(const u16x8*)(arow + kb + 8);
        v16u au;
        #pragma unroll
        for (int j = 0; j < 8; ++j) { au[j] = alo[j]; au[8 + j] = ahi[j]; }
        v16bf a = __builtin_bit_cast(v16bf, au);

        #pragma unroll
        for (int nt = 0; nt < NT; ++nt) {
            // B fragment: 32 contiguous bytes in LDS -> two ds_load_b128,
            // immediate offsets off a single per-lane base.
            const u16* bfrag = bbase + (size_t)(nt * NKB + kb32) * 32 * 16;
            u16x8 blo = *(const u16x8*)(bfrag);
            u16x8 bhi = *(const u16x8*)(bfrag + 8);
            v16u bu;
            #pragma unroll
            for (int j = 0; j < 8; ++j) { bu[j] = blo[j]; bu[8 + j] = bhi[j]; }
            v16bf b = __builtin_bit_cast(v16bf, bu);
            acc[nt] = __builtin_amdgcn_wmma_f32_16x16x32_bf16(
                /*neg_a=*/false, a, /*neg_b=*/false, b,
                /*c_mod=*/(short)0, acc[nt], /*reuse_a=*/false, /*reuse_b=*/false);
        }
    }
    // C/D layout: VGPR r -> M = half*8 + r, N = nt*16 + lrow
    float* __restrict__ cbase = C + (size_t)(wrow * 16 + half * 8) * N + lrow;
    #pragma unroll
    for (int nt = 0; nt < NT; ++nt)
        #pragma unroll
        for (int r = 0; r < 8; ++r)
            cbase[(size_t)r * N + nt * 16] = acc[nt][r];
}

// ---------------------------------------------------------------------------
// out[i,c] = bias[c] + dinv[i]^2 * h[i,c]   (self-loop term + bias)
// ---------------------------------------------------------------------------
__global__ void agg_init_kernel(const float* __restrict__ h,
                                const float* __restrict__ dinv,
                                const float* __restrict__ bias,
                                float* __restrict__ out,
                                int n, int C) {
    size_t i = (size_t)blockIdx.x * blockDim.x + threadIdx.x;
    if (i >= (size_t)n * C) return;
    int node = (int)(i / C);
    int c    = (int)(i % C);
    float di = dinv[node];
    out[i] = bias[c] + di * di * h[i];
}

// ---------------------------------------------------------------------------
// For each edge e: out[dst,:] += dinv[src]*ew*dinv[dst] * h[src,:]
// One thread per (edge, 4-channel group); float4 gather + 4x atomic f32 add.
// ---------------------------------------------------------------------------
__global__ void agg_edges_kernel(const int* __restrict__ src,
                                 const int* __restrict__ dst,
                                 const float* __restrict__ ew,
                                 const float* __restrict__ dinv,
                                 const float* __restrict__ h,
                                 float* __restrict__ out,
                                 int E, int C) {
    const int cg = C >> 2;
    size_t gid = (size_t)blockIdx.x * blockDim.x + threadIdx.x;
    if (gid >= (size_t)E * cg) return;
    const int e = (int)(gid / cg);
    const int g = (int)(gid % cg);
    const int s = src[e];
    const int d = dst[e];
    const float norm = dinv[s] * ew[e] * dinv[d];
    const float4 hv = *(const float4*)(h + (size_t)s * C + g * 4);
    float* o = out + (size_t)d * C + g * 4;
    atomicAdd(o + 0, norm * hv.x);
    atomicAdd(o + 1, norm * hv.y);
    atomicAdd(o + 2, norm * hv.z);
    atomicAdd(o + 3, norm * hv.w);
}

// ---------------------------------------------------------------------------
// Decode: out[e] = dot(z[a[e],:64], z[b[e],:64]); one wave32 per edge.
// ---------------------------------------------------------------------------
__global__ void decode_kernel(const float* __restrict__ z,
                              const int* __restrict__ eli,
                              float* __restrict__ out, int EL) {
    const int wave = blockIdx.x * (blockDim.x >> 5) + (threadIdx.x >> 5);
    if (wave >= EL) return;
    const int lane = threadIdx.x & 31;
    const int a = eli[wave];
    const int b = eli[EL + wave];
    const float* za = z + (size_t)a * 64;
    const float* zb = z + (size_t)b * 64;
    float p = za[lane] * zb[lane] + za[32 + lane] * zb[32 + lane];
    #pragma unroll
    for (int m = 16; m > 0; m >>= 1) p += __shfl_xor(p, m, 32);
    if (lane == 0) out[wave] = p;
}

// ---------------------------------------------------------------------------
// Host launcher
// ---------------------------------------------------------------------------
extern "C" void kernel_launch(void* const* d_in, const int* in_sizes, int n_in,
                              void* d_out, int out_size, void* d_ws, size_t ws_size,
                              hipStream_t stream) {
    const float* x   = (const float*)d_in[0];
    const int*   ei  = (const int*)  d_in[1];   // [2, E] -> src = ei, dst = ei + E
    const float* ew  = (const float*)d_in[2];
    const int*   eli = (const int*)  d_in[3];   // [2, EL]
    const float* W1  = (const float*)d_in[4];
    const float* b1  = (const float*)d_in[5];
    const float* W2  = (const float*)d_in[6];
    const float* b2  = (const float*)d_in[7];

    const int IN = 128, HID = 128, OUT = 64;
    const int N  = in_sizes[0] / IN;
    const int E  = in_sizes[1] / 2;
    const int EL = in_sizes[3] / 2;

    // Workspace layout (256B aligned), with buffer reuse.
    char*  ws  = (char*)d_ws;
    size_t off = 0;
    auto alloc = [&](size_t bytes) -> void* {
        void* p = ws + off;
        off = (off + bytes + 255) & ~(size_t)255;
        return p;
    };
    float* dinv = (float*)alloc((size_t)N * 4);
    u16*   xbf  = (u16*)  alloc((size_t)N * IN * 2);
    u16*   w1p  = (u16*)  alloc((size_t)IN * HID * 2);   // packed fragment-major
    u16*   w2p  = (u16*)  alloc((size_t)HID * OUT * 2);  // packed fragment-major
    float* h1   = (float*)alloc((size_t)N * HID * 4);    // GEMM1 out; reused as h2
    float* a1   = (float*)alloc((size_t)N * HID * 4);    // agg1 out; reused as z
    u16*   hbf  = xbf;   // relu(a1) in bf16 (xbf dead after GEMM1)
    float* h2   = h1;    // GEMM2 out (h1 dead after agg1 edges)
    float* z    = a1;    // final node embeddings (a1 dead after relu/cvt)

    const int T = 256;
    auto nb = [&](size_t n) -> int { return (int)((n + T - 1) / T); };

    // 1) degrees (self-loop weight 1.0) and dinv
    init_deg_kernel<<<nb(N), T, 0, stream>>>(dinv, N);
    accum_deg_kernel<<<nb(E), T, 0, stream>>>(ei + E, ew, dinv, E);
    finalize_dinv_kernel<<<nb(N), T, 0, stream>>>(dinv, N);

    // 2) bf16 conversion of x; pack W1, W2 into WMMA B-fragment layout
    f32_to_bf16_kernel<<<nb((size_t)N * IN), T, 0, stream>>>(x, xbf, (size_t)N * IN, 0);
    pack_B_kernel<<<nb((size_t)IN * HID), T, 0, stream>>>(W1, w1p, HID, IN);
    pack_B_kernel<<<nb((size_t)HID * OUT), T, 0, stream>>>(W2, w2p, OUT, HID);

    const int mtiles = N / 16;                 // M-tiles (M == #nodes)
    const int gblocks = (mtiles + 7) / 8;      // 8 waves per block

    // 3) GEMM1: h1 = x @ W1 (WMMA, 16x128 strip per wave, async-LDS B)
    gemm_bf16_wmma_kernel<128, 8><<<gblocks, 256, 0, stream>>>(xbf, w1p, h1, N);

    // 4) layer-1 aggregation: a1 = b1 + dinv^2*h1 (self loop) + edge scatter
    agg_init_kernel<<<nb((size_t)N * HID), T, 0, stream>>>(h1, dinv, b1, a1, N, HID);
    agg_edges_kernel<<<nb((size_t)E * (HID / 4)), T, 0, stream>>>(ei, ei + E, ew, dinv, h1, a1, E, HID);

    // 5) ReLU + bf16
    f32_to_bf16_kernel<<<nb((size_t)N * HID), T, 0, stream>>>(a1, hbf, (size_t)N * HID, 1);

    // 6) GEMM2: h2 = relu(h) @ W2 (16x64 strip per wave)
    gemm_bf16_wmma_kernel<128, 4><<<gblocks, 256, 0, stream>>>(hbf, w2p, h2, N);

    // 7) layer-2 aggregation: z = b2 + dinv^2*h2 + edge scatter
    agg_init_kernel<<<nb((size_t)N * OUT), T, 0, stream>>>(h2, dinv, b2, z, N, OUT);
    agg_edges_kernel<<<nb((size_t)E * (OUT / 4)), T, 0, stream>>>(ei, ei + E, ew, dinv, h2, z, E, OUT);

    // 8) decode: dot products over edge_label_index, one wave per edge
    decode_kernel<<<(EL + 7) / 8, 256, 0, stream>>>(z, eli, (float*)d_out, EL);
}
